// NeuralHawkesProcess_90847148245599
// MI455X (gfx1250) — matrix operand order repaired and checked by reading, not compiled, not run on hardware
//
#include <hip/hip_runtime.h>
#include <stdint.h>
#include <math.h>

// Problem dims (fixed by reference)
#define NB 256
#define NT 512
#define NH 512
#define NE 32
#define NG (5 * NH)       // 2560 gate columns
#define KSTEPS (NH / 32)  // 16 k-steps of 32 for K=512

typedef __attribute__((ext_vector_type(16))) __bf16 v16bf;
typedef __attribute__((ext_vector_type(8)))  float  v8f;

union FragBF { v16bf v; float4 f4[2]; };

__device__ __forceinline__ unsigned short f2bf(float x) {
  union { float f; unsigned u; } c; c.f = x;
  unsigned u = c.u;
  u += 0x7FFFu + ((u >> 16) & 1u);   // round-to-nearest-even
  return (unsigned short)(u >> 16);
}

// Native gfx1250 v_tanh_f32 when available (TANH is a CDNA5 TRANS32 op);
// falls back to libm otherwise.
__device__ __forceinline__ float fast_tanh(float x) {
#if __has_builtin(__builtin_amdgcn_tanhf)
  return __builtin_amdgcn_tanhf(x);
#else
  return tanhf(x);
#endif
}

// Branchless, numerically-stable softplus on native v_exp_f32/v_log_f32.
__device__ __forceinline__ float softplus_f(float x) {
  return fmaxf(x, 0.f) + __logf(1.f + __expf(-fabsf(x)));
}

// ---------------------------------------------------------------------------
// Precompute 1: xtable[e][j] = b_gates[j] + sum_k emb[e][k] * W_gates[k][j]
// ---------------------------------------------------------------------------
__global__ void nhp_build_xtable(const float* __restrict__ emb,
                                 const float* __restrict__ W_gates,
                                 const float* __restrict__ b_gates,
                                 float* __restrict__ xtable) {
  int id = blockIdx.x * blockDim.x + threadIdx.x;
  if (id >= NE * NG) return;
  int e = id / NG, j = id - e * NG;
  float acc = b_gates[j];
  for (int k = 0; k < NH; ++k)
    acc = fmaf(emb[e * NH + k], W_gates[(size_t)k * NG + j], acc);
  xtable[id] = acc;
}

// ---------------------------------------------------------------------------
// Precompute 2: pack W_h = W_gates[512:1024, :] into bf16 WMMA B-fragment
// order. Lane L: column N = L&15; lanes 0-15 = K 0..15, lanes 16-31 =
// K 16..31; dword j = K pair (2j, 2j+1) within the half.
// ---------------------------------------------------------------------------
__global__ void nhp_pack_wh(const float* __restrict__ W_gates,
                            unsigned* __restrict__ whpack) {
  int id = blockIdx.x * blockDim.x + threadIdx.x;
  const int total = (NG / 16) * KSTEPS * 32;   // 160*16*32
  if (id >= total) return;
  int lane = id & 31, frag = id >> 5;
  int ks = frag % KSTEPS, ntile = frag / KSTEPS;
  int n = lane & 15, khalf = lane >> 4;
#pragma unroll
  for (int j = 0; j < 8; ++j) {
    int k   = ks * 32 + khalf * 16 + 2 * j;
    int col = ntile * 16 + n;
    unsigned lo = f2bf(W_gates[(size_t)(NH + k)     * NG + col]);
    unsigned hi = f2bf(W_gates[(size_t)(NH + k + 1) * NG + col]);
    whpack[(size_t)id * 8 + j] = lo | (hi << 16);
  }
}

// Precompute 3: same packing for W_int [512 x 32]
__global__ void nhp_pack_wint(const float* __restrict__ W_int,
                              unsigned* __restrict__ wipack) {
  int id = blockIdx.x * blockDim.x + threadIdx.x;
  const int total = (NE / 16) * KSTEPS * 32;   // 1024
  if (id >= total) return;
  int lane = id & 31, frag = id >> 5;
  int ks = frag % KSTEPS, ntile = frag / KSTEPS;
  int n = lane & 15, khalf = lane >> 4;
#pragma unroll
  for (int j = 0; j < 8; ++j) {
    int k   = ks * 32 + khalf * 16 + 2 * j;
    int col = ntile * 16 + n;
    unsigned lo = f2bf(W_int[(size_t)k       * NE + col]);
    unsigned hi = f2bf(W_int[(size_t)(k + 1) * NE + col]);
    wipack[(size_t)id * 8 + j] = lo | (hi << 16);
  }
}

// A-fragment from LDS h (bf16 [16][512]): two ds_load_b128 per lane.
__device__ __forceinline__ v16bf load_a(const unsigned short (*h)[NH],
                                        int row, int khalf, int ks) {
  FragBF a;
  a.f4[0] = *(const float4*)&h[row][ks * 32 + khalf * 16];
  a.f4[1] = *(const float4*)&h[row][ks * 32 + khalf * 16 + 8];
  return a.v;
}

// B-fragment from pre-packed global buffer: two global_load_b128.
__device__ __forceinline__ v16bf load_b(const float4* __restrict__ pack,
                                        int fragIdx, int lane) {
  FragBF b;
  const float4* p = pack + ((size_t)fragIdx * 32 + lane) * 2;
  b.f4[0] = p[0];
  b.f4[1] = p[1];
  return b.v;
}

// ---------------------------------------------------------------------------
// One gate pass: 5-gate x 16-col GEMM slice (K=512) + LSTM cell update.
// `s` is a compile-time 0/1 at each call site so all register-array indices
// stay constant. Only 5 accumulators (40 VGPRs) live through the K loop.
// ---------------------------------------------------------------------------
__device__ __forceinline__ void gate_pass(
    const int s, const unsigned short (*h)[NH],
    const float4* __restrict__ whpack, const float* __restrict__ xtable,
    const int* ev_s, const float* dt_s,
    int w, int lane, int nloc, int khalf, int b0, int t,
    v8f& cfrag, v8f& hn, float* __restrict__ out_h) {
  v8f a0 = {}, a1 = {}, a2 = {}, a3 = {}, a4 = {};
#pragma unroll 1
  for (int ks = 0; ks < KSTEPS; ++ks) {
    v16bf a = load_a(h, nloc, khalf, ks);
    const int base = (w * 2 + s) * KSTEPS + ks;   // ntile = g*32 + w*2 + s
    v16bf b0f = load_b(whpack, base + 0 * 32 * KSTEPS, lane);
    v16bf b1f = load_b(whpack, base + 1 * 32 * KSTEPS, lane);
    v16bf b2f = load_b(whpack, base + 2 * 32 * KSTEPS, lane);
    v16bf b3f = load_b(whpack, base + 3 * 32 * KSTEPS, lane);
    v16bf b4f = load_b(whpack, base + 4 * 32 * KSTEPS, lane);
    a0 = __builtin_amdgcn_wmma_f32_16x16x32_bf16(false, a, false, b0f, (short)0, a0, false, false);
    a1 = __builtin_amdgcn_wmma_f32_16x16x32_bf16(false, a, false, b1f, (short)0, a1, false, false);
    a2 = __builtin_amdgcn_wmma_f32_16x16x32_bf16(false, a, false, b2f, (short)0, a2, false, false);
    a3 = __builtin_amdgcn_wmma_f32_16x16x32_bf16(false, a, false, b3f, (short)0, a3, false, false);
    a4 = __builtin_amdgcn_wmma_f32_16x16x32_bf16(false, a, false, b4f, (short)0, a4, false, false);
  }
  const int col = w * 32 + s * 16 + nloc;
#pragma unroll
  for (int v = 0; v < 8; ++v) {
    int   m  = v + 8 * khalf;
    int   ev = ev_s[m];
    float dt = dt_s[m];
    const float* xt = xtable + (size_t)ev * NG + col;
    float gi = a0[v] + xt[0 * NH];
    float gf = a1[v] + xt[1 * NH];
    float go = a2[v] + xt[2 * NH];
    float gc = a3[v] + xt[3 * NH];
    float gd = a4[v] + xt[4 * NH];
    float ig = fast_tanh(gi), fg = fast_tanh(gf);
    float og = fast_tanh(go), ct = fast_tanh(gc);
    float dec   = softplus_f(gd);
    float c_n   = fg * (cfrag[v] * __expf(-dec * dt)) + ig * ct;
    float h_n   = og * fast_tanh(c_n);
    cfrag[v] = c_n;
    hn[v]    = h_n;
    out_h[((size_t)(b0 + m) * NT + t) * NH + col] = h_n;
  }
}

// ---------------------------------------------------------------------------
// Main recurrent kernel: 16 WGs x 512 threads (16 wave32). WG owns batch rows
// [16*bx,16*bx+16); wave w owns hidden cols [32w,32w+32) across all 5 gates.
// Cell state lives in registers; only h round-trips through 16 KB of LDS.
// ---------------------------------------------------------------------------
__global__ __launch_bounds__(512, 1)
void nhp_recurrent_kernel(const int*   __restrict__ events,
                          const float* __restrict__ times,
                          const float* __restrict__ xtable,
                          const float4* __restrict__ whpack,
                          const float4* __restrict__ wipack,
                          const float* __restrict__ b_int,
                          float* __restrict__ out_lam,
                          float* __restrict__ out_h) {
  __shared__ __align__(16) unsigned short h_lds[16][NH];  // bf16 h, 16 KB
  __shared__ int   ev_s[16];
  __shared__ float dt_s[16];

  const int b0    = blockIdx.x * 16;
  const int tid   = threadIdx.x;
  const int w     = tid >> 5;     // wave id 0..15
  const int lane  = tid & 31;
  const int nloc  = lane & 15;
  const int khalf = lane >> 4;

  // h_0 = 0; stage ev/dt for t = 0
  for (int i = tid; i < 16 * NH; i += 512) ((unsigned short*)h_lds)[i] = 0;
  if (w == 15 && lane < 16) {
    int b = b0 + lane;
    ev_s[lane] = events[b * NT];
    dt_s[lane] = times[b * NT];
  }

  v8f c0 = {}, c1 = {};   // persistent cell state (rows v+8*khalf)
  v8f hn0, hn1;

  __syncthreads();

  for (int t = 0; t < NT; ++t) {
    // two sequential gate passes: read h_lds/ev_s/dt_s, update c, buffer hn
    gate_pass(0, h_lds, whpack, xtable, ev_s, dt_s,
              w, lane, nloc, khalf, b0, t, c0, hn0, out_h);
    gate_pass(1, h_lds, whpack, xtable, ev_s, dt_s,
              w, lane, nloc, khalf, b0, t, c1, hn1, out_h);

    __syncthreads();  // (a) all reads of h_{t-1}, ev/dt(t) complete

    // publish h_t to LDS; wave 15 stages next step's events / time deltas
#pragma unroll
    for (int v = 0; v < 8; ++v) {
      int m = v + 8 * khalf;
      h_lds[m][w * 32 + nloc]      = f2bf(hn0[v]);
      h_lds[m][w * 32 + 16 + nloc] = f2bf(hn1[v]);
    }
    if (w == 15 && lane < 16 && t + 1 < NT) {
      int b = b0 + lane;
      ev_s[lane] = events[b * NT + t + 1];
      dt_s[lane] = times[b * NT + t + 1] - times[b * NT + t];
    }

    __syncthreads();  // (b) h_t visible

    // intensity GEMM: lam = softplus(h_t @ W_int + b_int); 2 N-tiles
    if (w < 2) {  // wave-uniform branch, EXEC stays all-ones for WMMA
      v8f lacc = {};
#pragma unroll 1
      for (int ks = 0; ks < KSTEPS; ++ks) {
        v16bf a  = load_a(h_lds, nloc, khalf, ks);
        v16bf bf = load_b(wipack, w * KSTEPS + ks, lane);
        lacc = __builtin_amdgcn_wmma_f32_16x16x32_bf16(
            false, a, false, bf, (short)0, lacc, false, false);
      }
      int   e  = w * 16 + nloc;
      float be = b_int[e];
#pragma unroll
      for (int v = 0; v < 8; ++v) {
        int m = v + 8 * khalf;
        out_lam[((size_t)(b0 + m) * NT + t) * NE + e] = softplus_f(lacc[v] + be);
      }
    }
  }
}

// ---------------------------------------------------------------------------
extern "C" void kernel_launch(void* const* d_in, const int* in_sizes, int n_in,
                              void* d_out, int out_size, void* d_ws,
                              size_t ws_size, hipStream_t stream) {
  const int*   events  = (const int*)  d_in[0];  // [B,T]
  const float* times   = (const float*)d_in[1];  // [B,T]
  const float* emb     = (const float*)d_in[2];  // [E,H]
  const float* W_gates = (const float*)d_in[3];  // [2H,5H]
  const float* b_gates = (const float*)d_in[4];  // [5H]
  const float* W_int   = (const float*)d_in[5];  // [H,E]
  const float* b_int   = (const float*)d_in[6];  // [E]

  float* out_lam = (float*)d_out;                    // intensities [B,T,E]
  float* out_h   = out_lam + (size_t)NB * NT * NE;   // hidden      [B,T,H]

  char* ws = (char*)d_ws;
  float*    xtable = (float*)ws;                                   // 320 KB
  unsigned* whpack = (unsigned*)(ws + (size_t)NE * NG * 4);        // 2.5 MB
  unsigned* wipack = (unsigned*)(ws + (size_t)NE * NG * 4
                                    + (size_t)NG * NH * 2);        // 32 KB

  nhp_build_xtable<<<(NE * NG + 255) / 256, 256, 0, stream>>>(
      emb, W_gates, b_gates, xtable);
  nhp_pack_wh<<<((NG / 16) * KSTEPS * 32 + 255) / 256, 256, 0, stream>>>(
      W_gates, whpack);
  nhp_pack_wint<<<4, 256, 0, stream>>>(W_int, wipack);

  nhp_recurrent_kernel<<<NB / 16, 512, 0, stream>>>(
      events, times, xtable, (const float4*)whpack, (const float4*)wipack,
      b_int, out_lam, out_h);
}
